// QuantumHamiltonianLayer_83897891160833
// MI455X (gfx1250) — compile-verified, matrix operand order checked by reading, not run on hardware
//
#include <hip/hip_runtime.h>
#include <math.h>

typedef __attribute__((ext_vector_type(16))) _Float16 v16h;
typedef __attribute__((ext_vector_type(8)))  float    v8f;

#define NQ    10
#define NL    3
#define NT    128
#define DIM   1024
#define NPAIR 45
#define PI_F  3.14159265358979f

// ---------------------------------------------------------------------------
// Kernel 1: per-batch correlation matrix via v_wmma_f32_16x16x32_f16.
// One wave (32 lanes) per batch element. Lane l holds row r=l&15 of the
// normalized 10x128 matrix Xn; the K=128 reduction is 4 WMMAs of K=32.
// A and B operands are the same registers (Xn * Xn^T), which is exact
// regardless of per-lane K ordering since the K-sum is order invariant.
// Writes J_upper[45] = tanh(rho * exp(log_alpha)) per batch to workspace.
// ---------------------------------------------------------------------------
__global__ __launch_bounds__(256) void corr_kernel(
    const float* __restrict__ c, const float* __restrict__ dlt,
    const float* __restrict__ log_alpha, float* __restrict__ Jup, int nb)
{
  const int wave = threadIdx.x >> 5;
  const int lane = threadIdx.x & 31;
  const int b    = blockIdx.x * 8 + wave;
  if (b >= nb) return;                       // wave-uniform, safe around WMMA

  const int  r   = lane & 15;                // matrix row (M for A, N for B)
  const int  h   = lane >> 4;                // K half-group
  const bool act = (r < NQ);
  const float* cb = c   + (size_t)b * NT * NQ;
  const float* db = dlt + (size_t)b * NT * NQ;

  // prefetch next batch handled by this wave slot (global_prefetch_b8)
  if (b + 8 < nb) __builtin_prefetch(c + (size_t)(b + 8) * NT * NQ, 0, 1);

  // Per-lane 64 time samples following the documented 16-bit A 16x32 layout:
  // chunk j, element e -> K = 32j + e + (e>=8 ? 8 : 0) + 8h  (covers t=0..127)
  float f[64];
#pragma unroll
  for (int j = 0; j < 4; ++j) {
#pragma unroll
    for (int e = 0; e < 16; ++e) {
      const int t = 32 * j + e + (e >= 8 ? 8 : 0) + 8 * h;
      float v = 0.0f;
      if (act) v = cb[t * NQ + r] + 0.5f * db[t * NQ + r];
      f[j * 16 + e] = v;
    }
  }

  // mean / std over the full row (other half lives in lane ^ 16)
  float s1 = 0.0f;
#pragma unroll
  for (int i = 0; i < 64; ++i) s1 += f[i];
  const float mean = (s1 + __shfl_xor(s1, 16, 32)) * (1.0f / NT);

  float ss = 0.0f;
#pragma unroll
  for (int i = 0; i < 64; ++i) { f[i] -= mean; ss += f[i] * f[i]; }
  const float sst  = ss + __shfl_xor(ss, 16, 32);
  float stdv = sqrtf(sst * (1.0f / (NT - 1)));
  stdv = fmaxf(stdv, 1e-8f);
  const float inv = 1.0f / stdv;

  v16h a[4];
#pragma unroll
  for (int j = 0; j < 4; ++j)
#pragma unroll
    for (int e = 0; e < 16; ++e)
      a[j][e] = (_Float16)(f[j * 16 + e] * inv);

  v8f acc = {};
#pragma unroll
  for (int j = 0; j < 4; ++j)
    acc = __builtin_amdgcn_wmma_f32_16x16x32_f16(
        /*neg_a=*/false, a[j], /*neg_b=*/false, a[j],
        /*c_mod=*/(short)0, acc, /*reuse_a=*/false, /*reuse_b=*/false);

  // C layout: lane group h -> rows m = v + 8h, column n = r. Keep m < n < 10.
  const float alpha = expf(log_alpha[0]);
  float* out = Jup + (size_t)b * NPAIR;
  const int n = r;
#pragma unroll
  for (int v = 0; v < 8; ++v) {
    const int m = v + 8 * h;
    if (m < n && n < NQ) {
      const float rho = acc[v] * (1.0f / (NT - 1));
      const int   p   = m * (19 - m) / 2 + (n - m - 1);  // row-major triu index
      out[p] = tanhf(rho * alpha);
    }
  }
}

// ---------------------------------------------------------------------------
// Kernel 2: 10-qubit state-vector simulation. One block per batch element,
// 512 threads (16 wave32), state (1024 cfloat = 8 KB) resident in LDS.
// Diagonal gate layers (all RZs + entangling phase) are fused elementwise;
// each layer's 10 CNOTs collapse into one composite permutation gather.
// ---------------------------------------------------------------------------
__global__ __launch_bounds__(512) void sim_kernel(
    const float* __restrict__ c, const float* __restrict__ dlt,
    const float* __restrict__ vqc, const float* __restrict__ Jup,
    const float* __restrict__ w_proj, const float* __restrict__ b_proj,
    float* __restrict__ out)
{
  __shared__ float2 st[DIM];
  __shared__ float2 rot[NL][NQ][4];     // precomputed RZ*RY*RZ per (l,q)
  __shared__ float  theta[NQ], phi[NQ], Jl[NPAIR], wp[3 * NQ];
  __shared__ float  outAcc[4];

  const int b   = blockIdx.x;
  const int tid = threadIdx.x;

  if (tid < NQ) {
    theta[tid] = c  [((size_t)b * NT + (NT - 1)) * NQ + tid];
    phi[tid]   = dlt[((size_t)b * NT + (NT - 1)) * NQ + tid];
  }
  if (tid < NPAIR)  Jl[tid] = Jup[(size_t)b * NPAIR + tid];
  if (tid < 3 * NQ) wp[tid] = w_proj[tid];      // wp[a*10+q]
  if (tid < 4)      outAcc[tid] = 0.0f;
  if (tid >= 64 && tid < 64 + NL * NQ) {        // rot = rz(w2) @ ry(w1) @ rz(w0)
    const int i = tid - 64, l = i / NQ, q = i % NQ;
    const float w0 = vqc[(l * NQ + q) * 3 + 0];
    const float w1 = vqc[(l * NQ + q) * 3 + 1];
    const float w2 = vqc[(l * NQ + q) * 3 + 2];
    float cy, sy;  __sincosf(w1 * 0.5f, &sy, &cy);
    float sp, cp;  __sincosf((w2 + w0) * 0.5f, &sp, &cp);
    float sm, cm;  __sincosf((w2 - w0) * 0.5f, &sm, &cm);
    rot[l][q][0] = make_float2( cy * cp, -cy * sp);   // g00 = cy e^{-i(w2+w0)/2}
    rot[l][q][1] = make_float2(-sy * cm,  sy * sm);   // g01 = -sy e^{-i(w2-w0)/2}
    rot[l][q][2] = make_float2( sy * cm,  sy * sm);   // g10 =  sy e^{+i(w2-w0)/2}
    rot[l][q][3] = make_float2( cy * cp,  cy * sp);   // g11 = cy e^{+i(w2+w0)/2}
  }
  st[tid]       = make_float2(0.0f, 0.0f);
  st[tid + 512] = make_float2(0.0f, 0.0f);
  __syncthreads();
  if (tid == 0) st[0] = make_float2(1.0f, 0.0f);
  __syncthreads();

  // paired 2x2 gate on bit m (qubit q acts on bit 9-q); pairs are disjoint per
  // thread so only the trailing barrier is required.
  auto gate2 = [&](int m, float2 g00, float2 g01, float2 g10, float2 g11) {
    const unsigned p  = (unsigned)tid;
    const unsigned i0 = ((p >> m) << (m + 1)) | (p & ((1u << m) - 1u));
    const unsigned i1 = i0 | (1u << m);
    const float2 a0 = st[i0], a1 = st[i1];
    st[i0] = make_float2(g00.x * a0.x - g00.y * a0.y + g01.x * a1.x - g01.y * a1.y,
                         g00.x * a0.y + g00.y * a0.x + g01.x * a1.y + g01.y * a1.x);
    st[i1] = make_float2(g10.x * a0.x - g10.y * a0.y + g11.x * a1.x - g11.y * a1.y,
                         g10.x * a0.y + g10.y * a0.x + g11.x * a1.y + g11.y * a1.x);
    __syncthreads();
  };

  // --- initial RY(theta_q) layer (RZ part deferred into fused diagonal) ---
  for (int q = 0; q < NQ; ++q) {
    float sy, cy;  __sincosf(theta[q] * 0.5f, &sy, &cy);
    gate2(9 - q, make_float2(cy, 0.f), make_float2(-sy, 0.f),
                 make_float2(sy, 0.f), make_float2(cy, 0.f));
  }

  // --- fused diagonal: all RZ(phi_q) + entangling phase exp(i*pi/2*J.masks) ---
  for (int k = 0; k < 2; ++k) {
    const unsigned s = (unsigned)tid + k * 512u;
    float hang = 0.0f;
    for (int q = 0; q < NQ; ++q)
      hang += (((s >> (9 - q)) & 1u) ? 0.5f : -0.5f) * phi[q];
    int p = 0;
    for (int i = 0; i < NQ; ++i)
      for (int j = i + 1; j < NQ; ++j, ++p) {
        const float sgn = ((((s >> (9 - i)) ^ (s >> (9 - j))) & 1u) ? 1.0f : -1.0f);
        hang += (0.5f * PI_F) * Jl[p] * sgn;
      }
    float sn, cs;  __sincosf(hang, &sn, &cs);
    const float2 a = st[s];
    st[s] = make_float2(a.x * cs - a.y * sn, a.x * sn + a.y * cs);
  }
  __syncthreads();

  // --- VQC layers ---
  for (int l = 0; l < NL; ++l) {
    for (int q = 0; q < NQ; ++q)
      gate2(9 - q, rot[l][q][0], rot[l][q][1], rot[l][q][2], rot[l][q][3]);

    // composite CNOT permutation: new[s] = old[P0(P1(...P9(s)))]
    const int rng = l + 1;                 // l % (NQ-1) + 1
    float2 v0, v1;
    {
      unsigned x = (unsigned)tid;
      for (int q = 9; q >= 0; --q) {
        const int tgt = (q + rng) % NQ;
        if ((x >> (9 - q)) & 1u) x ^= (1u << (9 - tgt));
      }
      v0 = st[x];
    }
    {
      unsigned x = (unsigned)tid + 512u;
      for (int q = 9; q >= 0; --q) {
        const int tgt = (q + rng) % NQ;
        if ((x >> (9 - q)) & 1u) x ^= (1u << (9 - tgt));
      }
      v1 = st[x];
    }
    __syncthreads();
    st[tid] = v0;  st[tid + 512] = v1;
    __syncthreads();

    if (l < NL - 1) {
      for (int q = 0; q < NQ; ++q) {       // RY(theta_q * 0.5)
        float sy, cy;  __sincosf(theta[q] * 0.25f, &sy, &cy);
        gate2(9 - q, make_float2(cy, 0.f), make_float2(-sy, 0.f),
                     make_float2(sy, 0.f), make_float2(cy, 0.f));
      }
      for (int k = 0; k < 2; ++k) {        // fused RZ(phi_q * 0.5)
        const unsigned s = (unsigned)tid + k * 512u;
        float hang = 0.0f;
        for (int q = 0; q < NQ; ++q)
          hang += (((s >> (9 - q)) & 1u) ? 0.25f : -0.25f) * phi[q];
        float sn, cs;  __sincosf(hang, &sn, &cs);
        const float2 a = st[s];
        st[s] = make_float2(a.x * cs - a.y * sn, a.x * sn + a.y * cs);
      }
      __syncthreads();
    }
  }

  // --- probs -> (w_proj . ZSIGNS)-weighted reduction, 3 scalars per block ---
  float acc0 = 0.f, acc1 = 0.f, acc2 = 0.f;
  for (int k = 0; k < 2; ++k) {
    const unsigned s = (unsigned)tid + k * 512u;
    const float2 a = st[s];
    const float pr = a.x * a.x + a.y * a.y;
    float w0 = 0.f, w1 = 0.f, w2 = 0.f;
    for (int q = 0; q < NQ; ++q) {
      const float sgn = (((s >> (9 - q)) & 1u) ? -1.0f : 1.0f);
      w0 += wp[q] * sgn;  w1 += wp[NQ + q] * sgn;  w2 += wp[2 * NQ + q] * sgn;
    }
    acc0 += pr * w0;  acc1 += pr * w1;  acc2 += pr * w2;
  }
  for (int off = 16; off > 0; off >>= 1) {
    acc0 += __shfl_xor(acc0, off, 32);
    acc1 += __shfl_xor(acc1, off, 32);
    acc2 += __shfl_xor(acc2, off, 32);
  }
  if ((tid & 31) == 0) {                    // one ds_add_f32 per wave
    atomicAdd(&outAcc[0], acc0);
    atomicAdd(&outAcc[1], acc1);
    atomicAdd(&outAcc[2], acc2);
  }
  __syncthreads();
  if (tid < 3) out[b * 3 + tid] = outAcc[tid] + b_proj[tid];
}

// ---------------------------------------------------------------------------
extern "C" void kernel_launch(void* const* d_in, const int* in_sizes, int n_in,
                              void* d_out, int out_size, void* d_ws, size_t ws_size,
                              hipStream_t stream) {
  const float* c   = (const float*)d_in[0];   // c_kt       (B,T,10)
  const float* dlt = (const float*)d_in[1];   // delta_c_kt (B,T,10)
  const float* vqc = (const float*)d_in[2];   // vqc_weights (3,10,3)
  const float* wpj = (const float*)d_in[3];   // w_proj (3,10)
  const float* bpj = (const float*)d_in[4];   // b_proj (3,)
  const float* la  = (const float*)d_in[5];   // log_alpha scalar

  const int nb = in_sizes[0] / (NT * NQ);     // B = 4096
  float* Jup = (float*)d_ws;                  // nb * 45 floats

  corr_kernel<<<(nb + 7) / 8, 256, 0, stream>>>(c, dlt, la, Jup, nb);
  sim_kernel<<<nb, 512, 0, stream>>>(c, dlt, vqc, Jup, wpj, bpj, (float*)d_out);
}
